// MoE_65601330479265
// MI455X (gfx1250) — matrix operand order repaired and checked by reading, not compiled, not run on hardware
//
#include <hip/hip_runtime.h>
#include <hip/hip_bf16.h>
#include <stdint.h>

// Problem constants (match reference)
#define N_TOK   32768
#define D_IN    1024
#define D_OUT   1024
#define NE      8
#define LN_EPS  1e-5f

// Tiling
#define M_TILE      32          // bucket rows per block
#define LDS_STRIDE  1028        // floats per LDS row (1028%64==4 -> conflict-free halves)

typedef __attribute__((ext_vector_type(16))) __bf16 v16bf;
typedef __attribute__((ext_vector_type(8)))  float  v8f;

union Frag {
    struct { uint4 lo, hi; } u;   // 32 bytes
    v16bf v;
};

__device__ __forceinline__ unsigned short f32_to_bf16(float f) {
    union { float f; unsigned int u; } c; c.f = f;
    unsigned int u = c.u;
    unsigned int r = u + 0x7FFFu + ((u >> 16) & 1u);   // round-to-nearest-even
    return (unsigned short)(r >> 16);
}

__device__ __forceinline__ float gelu_exact(float h) {
    return 0.5f * h * (1.0f + erff(h * 0.70710678118654752440f));
}

// A fragment (16x32 bf16): elems [0..7] at +off, [8..15] at +off+16 (K halves)
__device__ __forceinline__ void load_a(Frag& a, const unsigned short* p, int off) {
    a.u.lo = *(const uint4*)(p + off);
    a.u.hi = *(const uint4*)(p + off + 16);
}
// B fragments for all 8 column tiles; tile stride 16*D_IN folds into imm offset
__device__ __forceinline__ void load_b(Frag* b, const unsigned short* p, int off) {
#pragma unroll
    for (int ct = 0; ct < 8; ++ct) {
        b[ct].u.lo = *(const uint4*)(p + off + ct * (16 * D_IN));
        b[ct].u.hi = *(const uint4*)(p + off + ct * (16 * D_IN) + 16);
    }
}

// ---------------- zero output + counters ----------------
__global__ void zero_kernel(float* out, int* counts, long n) {
    long i = (long)blockIdx.x * blockDim.x + threadIdx.x;
    long stride = (long)gridDim.x * blockDim.x;
    for (long j = i; j < n; j += stride) out[j] = 0.0f;
    if (i < NE) counts[i] = 0;
}

// ---------------- x (f32) -> x (bf16) ----------------
__global__ void cvt_x_kernel(const float* __restrict__ x,
                             unsigned short* __restrict__ xb, long n) {
    long i = (long)blockIdx.x * blockDim.x + threadIdx.x;
    long stride = (long)gridDim.x * blockDim.x;
    for (long j = i; j < n; j += stride) xb[j] = f32_to_bf16(x[j]);
}

// ---------------- We [e][i][o] f32 -> Bt [e][o][i] bf16 (transposed) ----------------
__global__ void cvt_w_kernel(const float* __restrict__ We,
                             unsigned short* __restrict__ Bt) {
    long n = (long)NE * D_IN * D_OUT;
    long i = (long)blockIdx.x * blockDim.x + threadIdx.x;
    long stride = (long)gridDim.x * blockDim.x;
    for (long idx = i; idx < n; idx += stride) {
        int o = (int)(idx % D_OUT);
        long t = idx / D_OUT;
        int ii = (int)(t % D_IN);
        int e  = (int)(t / D_IN);
        Bt[((long)(e * D_OUT + o)) * D_IN + ii] = f32_to_bf16(We[idx]);
    }
}

// ---------------- gating: one wave32 per token ----------------
__global__ void gate_kernel(const float* __restrict__ x,
                            const float* __restrict__ Wg,
                            const float* __restrict__ bg,
                            int* counts, int* bucket, float* bgate) {
    int lane = threadIdx.x & 31;
    int wave = threadIdx.x >> 5;
    int n = blockIdx.x * 8 + wave;
    if (n >= N_TOK) return;

    float acc[NE];
#pragma unroll
    for (int e = 0; e < NE; ++e) acc[e] = 0.0f;

    const float* xr = x + (long)n * D_IN;
    for (int i = lane; i < D_IN; i += 32) {
        float xv = xr[i];
        const float* wr = Wg + i * NE;
#pragma unroll
        for (int e = 0; e < NE; ++e) acc[e] += xv * wr[e];
    }
#pragma unroll
    for (int m = 16; m >= 1; m >>= 1) {
#pragma unroll
        for (int e = 0; e < NE; ++e) acc[e] += __shfl_xor(acc[e], m, 32);
    }

    if (lane == 0) {
        float lg[NE];
#pragma unroll
        for (int e = 0; e < NE; ++e) lg[e] = tanhf(acc[e] + bg[e]);
        // top-2 (lowest index wins ties, like jax top_k)
        int i0 = 0; float v0 = lg[0];
#pragma unroll
        for (int e = 1; e < NE; ++e) if (lg[e] > v0) { v0 = lg[e]; i0 = e; }
        int i1 = -1; float v1 = -3.4e38f;
#pragma unroll
        for (int e = 0; e < NE; ++e) if (e != i0 && lg[e] > v1) { v1 = lg[e]; i1 = e; }
        // softmax over the two gates (v0 >= v1)
        float ex = expf(v1 - v0);
        float inv = 1.0f / (1.0f + ex);
        float g0 = inv, g1 = ex * inv;

        int p0 = atomicAdd(&counts[i0], 1);
        bucket[i0 * N_TOK + p0] = n;  bgate[i0 * N_TOK + p0] = g0;
        int p1 = atomicAdd(&counts[i1], 1);
        bucket[i1 * N_TOK + p1] = n;  bgate[i1 * N_TOK + p1] = g1;
    }
}

// ---------------- grouped expert GEMM + GELU + LN + combine ----------------
// grid.x = NE * (N_TOK / M_TILE); block = 512 (16 waves)
// Block computes a 32-row x 1024-col tile for one expert's bucket slice.
// Per wave: 16 rows x 128 cols = 8 WMMA accumulators; A+B double-buffered so a
// full s_clause load batch for k+32 is in flight while the 8 WMMAs for k run.
__global__ void __launch_bounds__(512, 1)
expert_kernel(const unsigned short* __restrict__ xb,
              const unsigned short* __restrict__ Bt,
              const float* __restrict__ be,
              const float* __restrict__ gamma,
              const float* __restrict__ beta,
              const int* __restrict__ counts,
              const int* __restrict__ bucket,
              const float* __restrict__ bgate,
              float* __restrict__ out) {
    extern __shared__ float hsm[];                 // [M_TILE][LDS_STRIDE]
    __shared__ int   s_tok[M_TILE];
    __shared__ float s_gate[M_TILE];

    int e  = blockIdx.x >> 10;                     // 1024 tiles per expert
    int tt = blockIdx.x & 1023;
    int cnt = counts[e];
    int base = tt * M_TILE;
    if (base >= cnt) return;
    int rows = min(M_TILE, cnt - base);

    int tid = threadIdx.x;
    if (tid < M_TILE) {
        int idx = base + tid;
        int cl = idx < cnt ? idx : cnt - 1;        // clamp padded rows to a valid token
        s_tok[tid]  = bucket[e * N_TOK + cl];
        s_gate[tid] = (idx < cnt) ? bgate[e * N_TOK + idx] : 0.0f;
    }
    __syncthreads();

    int lane   = tid & 31;
    int wave   = tid >> 5;                         // 0..15
    int rsub   = wave & 1;                         // row sub-tile: rows 0-15 / 16-31
    int coct   = wave >> 1;                        // 0..7 -> 128-col strip
    int rbase  = rsub * 16;
    int cbase  = coct * 128;
    int m      = lane & 15;
    int hiHalf = lane >> 4;

    int tokenRow = s_tok[rbase + m];

    const unsigned short* aPtr = xb + (long)tokenRow * D_IN + (hiHalf << 3);
    const unsigned short* bPtr =
        Bt + ((long)(e * D_OUT + cbase + m)) * D_IN + (hiHalf << 3);

    v8f zero = {};
    v8f acc[8];
#pragma unroll
    for (int ct = 0; ct < 8; ++ct) acc[ct] = zero;

    // Software pipeline: two stages of (A frag + 8 B frags)
    Frag a0, a1;
    Frag b0[8], b1[8];
    load_a(a0, aPtr, 0);
    load_b(b0, bPtr, 0);

    for (int k0 = 0; k0 < D_IN; k0 += 64) {
        // prefetch stage 1 @ k0+32
        load_a(a1, aPtr, 32);
        load_b(b1, bPtr, 32);
        // compute stage 0 @ k0
#pragma unroll
        for (int ct = 0; ct < 8; ++ct)
            acc[ct] = __builtin_amdgcn_wmma_f32_16x16x32_bf16(
                false, a0.v, false, b0[ct].v, (short)0, acc[ct], false, false);
        aPtr += 64;
        bPtr += 64;
        // prefetch stage 0 @ k0+64 (skip past end)
        if (k0 + 64 < D_IN) {
            load_a(a0, aPtr, 0);
            load_b(b0, bPtr, 0);
        }
        // compute stage 1 @ k0+32
#pragma unroll
        for (int ct = 0; ct < 8; ++ct)
            acc[ct] = __builtin_amdgcn_wmma_f32_16x16x32_bf16(
                false, a1.v, false, b1[ct].v, (short)0, acc[ct], false, false);
    }

    // C layout: lanes 0-15 vgpr v -> (M=v, N=lane); lanes 16-31 -> (M=v+8, N=lane-16)
#pragma unroll
    for (int ct = 0; ct < 8; ++ct) {
        int col = cbase + ct * 16 + m;
        float bias = be[e * D_OUT + col];
#pragma unroll
        for (int v = 0; v < 8; ++v) {
            int row = rbase + v + (hiHalf ? 8 : 0);
            float h = acc[ct][v] + bias;
            hsm[row * LDS_STRIDE + col] = gelu_exact(h);
        }
    }
    __syncthreads();

    // LayerNorm + gate-weighted combine: 2 rows per wave (16 waves x 2 = 32)
    for (int rr = 0; rr < 2; ++rr) {
        int row = wave * 2 + rr;
        if (row >= rows) continue;
        const float* hr = hsm + row * LDS_STRIDE;
        float s = 0.0f, sq = 0.0f;
        for (int c = lane; c < D_OUT; c += 32) {
            float v = hr[c];
            s += v; sq += v * v;
        }
#pragma unroll
        for (int mm = 16; mm >= 1; mm >>= 1) {
            s  += __shfl_xor(s,  mm, 32);
            sq += __shfl_xor(sq, mm, 32);
        }
        float mean = s * (1.0f / D_OUT);
        float var  = sq * (1.0f / D_OUT) - mean * mean;
        float rstd = rsqrtf(var + LN_EPS);
        int   token = s_tok[row];
        float g     = s_gate[row];
        float* orow = out + (long)token * D_OUT;
        for (int c = lane; c < D_OUT; c += 32) {
            float v = (hr[c] - mean) * rstd;
            v = v * gamma[e * D_OUT + c] + beta[e * D_OUT + c];
            atomicAdd(&orow[c], g * v);   // exactly 2 adds per element -> deterministic
        }
    }
}

extern "C" void kernel_launch(void* const* d_in, const int* in_sizes, int n_in,
                              void* d_out, int out_size, void* d_ws, size_t ws_size,
                              hipStream_t stream) {
    const float* x     = (const float*)d_in[0];
    const float* Wg    = (const float*)d_in[1];
    const float* bg    = (const float*)d_in[2];
    const float* We    = (const float*)d_in[3];
    const float* be    = (const float*)d_in[4];
    const float* gamma = (const float*)d_in[5];
    const float* beta  = (const float*)d_in[6];
    float* out = (float*)d_out;

    // Workspace carve-up (~86 MB total)
    char* ws = (char*)d_ws;
    size_t off = 0;
    unsigned short* xb = (unsigned short*)(ws + off); off += (size_t)N_TOK * D_IN * 2;          // 64 MB
    unsigned short* Bt = (unsigned short*)(ws + off); off += (size_t)NE * D_IN * D_OUT * 2;     // 16 MB
    int*   counts = (int*)(ws + off);   off += 256;
    int*   bucket = (int*)(ws + off);   off += (size_t)NE * N_TOK * 4;                          // 1 MB
    float* bgate  = (float*)(ws + off); off += (size_t)NE * N_TOK * 4;                          // 1 MB
    (void)ws_size; (void)in_sizes; (void)n_in; (void)out_size;

    zero_kernel<<<4096, 256, 0, stream>>>(out, counts, (long)N_TOK * D_OUT);
    cvt_x_kernel<<<4096, 256, 0, stream>>>(x, xb, (long)N_TOK * D_IN);
    cvt_w_kernel<<<4096, 256, 0, stream>>>(We, Bt);
    gate_kernel<<<N_TOK / 8, 256, 0, stream>>>(x, Wg, bg, counts, bucket, bgate);

    size_t smem = (size_t)M_TILE * LDS_STRIDE * sizeof(float);   // ~131 KB (< 320 KB/WGP)
    expert_kernel<<<NE * (N_TOK / M_TILE), 512, smem, stream>>>(
        xb, Bt, be, gamma, beta, counts, bucket, bgate, out);
}
